// _InteractionBlock_48782238548371
// MI455X (gfx1250) — matrix operand order repaired and checked by reading, not compiled, not run on hardware
//
#include <hip/hip_runtime.h>

// CDNA5 / gfx1250: wave32, WMMA 16x16x32 bf16, f32 accumulate.
// All LDS operands are stored pre-swizzled into WMMA fragment order so each
// lane's A/B fragment is 32 contiguous bytes (2x ds_load_b128, no packing movs).
typedef __bf16 v16bf __attribute__((ext_vector_type(16)));
typedef float  v8f   __attribute__((ext_vector_type(8)));

#define D_MODEL 128
#define N_RBF   50
#define LN2F    0.69314718055994530942f
#define LOG2EF  1.44269504088896340736f

// shifted softplus on the hardware transcendental pipe (v_exp_f32/v_log_f32):
//   ssp(x) = max(x,0) + ln2*log2(1 + 2^(-|x|*log2e)) - ln2
// branch-free, numerically stable for large |x|.
__device__ __forceinline__ float ssp(float v) {
    float t = __builtin_amdgcn_exp2f(-fabsf(v) * LOG2EF);
    return fmaxf(v, 0.0f) + LN2F * (__builtin_amdgcn_logf(1.0f + t) - 1.0f);
}

// Permutation of k within each 32-wide K-block such that the 16 A-matrix
// elements of a lane (K = {k0+8h..k0+8h+7} U {k0+16+8h..+7}, per ISA 7.12.2)
// are contiguous at [k0 + 16*half, k0 + 16*half + 16).
__device__ __forceinline__ int swz32(int k) {
    return (k & ~31) + (((k >> 3) & 1) << 4) + (k & 7) + (((k >> 4) & 1) << 3);
}

__device__ __forceinline__ void atomic_add_f32(float* p, float v) {
    __hip_atomic_fetch_add(p, v, __ATOMIC_RELAXED, __HIP_MEMORY_SCOPE_AGENT);
}

__device__ __forceinline__ v8f wmma_bf16(v16bf a, v16bf b, v8f c) {
    return __builtin_amdgcn_wmma_f32_16x16x32_bf16(false, a, false, b,
                                                   (short)0, c, false, false);
}

// Cooperative fill of a B-weight LDS buffer in fragment-major order:
// sW[(ks*8+t)*32*16 + lane*16 + e] = W[ks*32 + 8*(lane>>4) + (e&7) + 16*(e>>3)][t*16 + (lane&15)]
__device__ __forceinline__ void fill_b_swizzled(__bf16* sW, const float* __restrict__ W,
                                                int ksteps, int krows, int tid) {
    const int total = ksteps * 8 * 32 * 16;
    for (int i = tid; i < total; i += 128) {
        int e  = i & 15;
        int ln = (i >> 4) & 31;
        int t  = (i >> 9) & 7;
        int ks = i >> 12;
        int K   = ks * 32 + 8 * (ln >> 4) + (e & 7) + 16 * (e >> 3);
        int col = t * 16 + (ln & 15);
        sW[i] = (K < krows) ? (__bf16)W[K * D_MODEL + col] : (__bf16)0.0f;
    }
}

// ---------------------------------------------------------------------------
// Edge kernel: per 16-edge tile per wave
//   h = ssp(e_ij @ fW1 + fb1); W = h @ fW2 + fb2; atomicAdd(m[src], W*x[dst])
// LDS: sW1 16KB + sW2 32KB + per-wave h tile 16KB = 65536 B.
// The 16x64 e_ij staging tile aliases the front of the per-wave h tile
// (stage-1 reads finish before h writes; same-wave DS ops are in-order).
// ---------------------------------------------------------------------------
__global__ __launch_bounds__(128)
void edge_filter_scatter_kernel(const float* __restrict__ x,
                                const int*   __restrict__ eidx,   // [2][E]
                                const float* __restrict__ eij,    // [E][50]
                                const float* __restrict__ fW1,    // [50][128]
                                const float* __restrict__ fb1,
                                const float* __restrict__ fW2,    // [128][128]
                                const float* __restrict__ fb2,
                                float* __restrict__ m,            // [N][128]
                                int N, int E)
{
    __shared__ __bf16 sW1[2 * 8 * 32 * 16];       // 16 KB, fragment-major
    __shared__ __bf16 sW2[4 * 8 * 32 * 16];       // 32 KB, fragment-major
    __shared__ __bf16 sH [4 * 16 * D_MODEL];      // 16 KB, per-wave h (swz cols)

    const int tid  = threadIdx.x;
    const int wave = tid >> 5;
    const int lane = tid & 31;
    const int half = lane >> 4;
    const int n16  = lane & 15;

    fill_b_swizzled(sW1, fW1, 2, N_RBF, tid);
    fill_b_swizzled(sW2, fW2, 4, D_MODEL, tid);
    __syncthreads();

    const int ebase = blockIdx.x * 64 + wave * 16;
    __bf16* myH = &sH[wave * 16 * D_MODEL];
    __bf16* myE = myH;                            // aliased 16x64 staging

    // zero-pad staging (covers K=50..63), then coalesced fill with swz cols
    {
        unsigned long long* z = (unsigned long long*)myE;
        for (int i = lane; i < (16 * 64) / 4; i += 32) z[i] = 0ull;
    }
    for (int i = lane; i < 16 * N_RBF; i += 32) {
        int row  = i / N_RBF;
        int k    = i - row * N_RBF;
        int edge = ebase + row;
        float v  = (edge < E) ? eij[(size_t)edge * N_RBF + k] : 0.0f;
        myE[row * 64 + swz32(k)] = (__bf16)v;
    }

    // ---- stage 1: h = ssp(eij @ fW1 + fb1) ----
    v8f acc[8] = {};
    {
        const v16bf* B1 = (const v16bf*)sW1;
#pragma unroll
        for (int ks = 0; ks < 2; ++ks) {
            v16bf a = *(const v16bf*)&myE[n16 * 64 + ks * 32 + 16 * half];
#pragma unroll
            for (int t = 0; t < 8; ++t)
                acc[t] = wmma_bf16(a, B1[(ks * 8 + t) * 32 + lane], acc[t]);
        }
    }

    // bias + ssp; write h with swizzled columns for stage-2 A fragments
#pragma unroll
    for (int t = 0; t < 8; ++t) {
        int col = t * 16 + n16;
        int sc  = swz32(col);
        float bias = fb1[col];
#pragma unroll
        for (int r = 0; r < 8; ++r)
            myH[(r + 8 * half) * D_MODEL + sc] = (__bf16)ssp(acc[t][r] + bias);
    }

    // ---- stage 2: W = h @ fW2 + fb2 ----
    v8f acc2[8] = {};
    {
        const v16bf* B2 = (const v16bf*)sW2;
#pragma unroll
        for (int ks = 0; ks < 4; ++ks) {
            v16bf a = *(const v16bf*)&myH[n16 * D_MODEL + ks * 32 + 16 * half];
#pragma unroll
            for (int t = 0; t < 8; ++t)
                acc2[t] = wmma_bf16(a, B2[(ks * 8 + t) * 32 + lane], acc2[t]);
        }
    }

    // ---- fused gather x[dst] * W, scatter-add into m[src] (L2-resident) ----
#pragma unroll
    for (int r = 0; r < 8; ++r) {
        int edge = ebase + r + 8 * half;          // C/D layout: M = r + 8*half
        if (edge >= E) continue;
        int src = eidx[edge];
        int dst = eidx[E + edge];
        const float* xr = x + (size_t)dst * D_MODEL;
        float*       mr = m + (size_t)src * D_MODEL;
#pragma unroll
        for (int t = 0; t < 8; ++t) {
            int col = t * 16 + n16;
            float w = acc2[t][r] + fb2[col];
            atomic_add_f32(&mr[col], w * xr[col]);
        }
    }
}

// ---------------------------------------------------------------------------
// Node kernel: u = ssp(m @ uW1 + ub1); out = x + u @ uW2 + ub2
// LDS: one 32KB fragment-major weight buffer (uW1 then uW2) + 16KB staging.
// ---------------------------------------------------------------------------
__global__ __launch_bounds__(128)
void node_update_kernel(const float* __restrict__ x,
                        const float* __restrict__ m,
                        const float* __restrict__ uW1, const float* __restrict__ ub1,
                        const float* __restrict__ uW2, const float* __restrict__ ub2,
                        float* __restrict__ out, int N)
{
    __shared__ __bf16 sW[4 * 8 * 32 * 16];        // 32 KB
    __shared__ __bf16 sS[4 * 16 * D_MODEL];       // 16 KB (swz cols)

    const int tid  = threadIdx.x;
    const int wave = tid >> 5;
    const int lane = tid & 31;
    const int half = lane >> 4;
    const int n16  = lane & 15;

    const int nbase  = blockIdx.x * 64;
    const int nwbase = nbase + wave * 16;
    __bf16* myS = &sS[wave * 16 * D_MODEL];

    fill_b_swizzled(sW, uW1, 4, D_MODEL, tid);
    for (int i = tid; i < 64 * D_MODEL; i += 128) {
        int row  = i >> 7;
        int k    = i & 127;
        int node = nbase + row;
        float v  = (node < N) ? m[(size_t)node * D_MODEL + k] : 0.0f;
        sS[row * D_MODEL + swz32(k)] = (__bf16)v;
    }
    __syncthreads();

    // ---- stage 1: u = ssp(m @ uW1 + ub1) ----
    v8f acc[8] = {};
    {
        const v16bf* B = (const v16bf*)sW;
#pragma unroll
        for (int ks = 0; ks < 4; ++ks) {
            v16bf a = *(const v16bf*)&myS[n16 * D_MODEL + ks * 32 + 16 * half];
#pragma unroll
            for (int t = 0; t < 8; ++t)
                acc[t] = wmma_bf16(a, B[(ks * 8 + t) * 32 + lane], acc[t]);
        }
    }
    __syncthreads();                              // done reading uW1 / m tile

    // stage u (swz cols); swap weight buffer to uW2
#pragma unroll
    for (int t = 0; t < 8; ++t) {
        int col = t * 16 + n16;
        int sc  = swz32(col);
        float bias = ub1[col];
#pragma unroll
        for (int r = 0; r < 8; ++r)
            myS[(r + 8 * half) * D_MODEL + sc] = (__bf16)ssp(acc[t][r] + bias);
    }
    fill_b_swizzled(sW, uW2, 4, D_MODEL, tid);
    __syncthreads();

    // ---- stage 2: out = x + u @ uW2 + ub2 ----
    v8f acc2[8] = {};
    {
        const v16bf* B = (const v16bf*)sW;
#pragma unroll
        for (int ks = 0; ks < 4; ++ks) {
            v16bf a = *(const v16bf*)&myS[n16 * D_MODEL + ks * 32 + 16 * half];
#pragma unroll
            for (int t = 0; t < 8; ++t)
                acc2[t] = wmma_bf16(a, B[(ks * 8 + t) * 32 + lane], acc2[t]);
        }
    }

#pragma unroll
    for (int r = 0; r < 8; ++r) {
        int node = nwbase + r + 8 * half;
        if (node >= N) continue;
#pragma unroll
        for (int t = 0; t < 8; ++t) {
            int col = t * 16 + n16;
            size_t idx = (size_t)node * D_MODEL + col;
            out[idx] = x[idx] + acc2[t][r] + ub2[col];
        }
    }
}

// ---------------------------------------------------------------------------
extern "C" void kernel_launch(void* const* d_in, const int* in_sizes, int n_in,
                              void* d_out, int out_size, void* d_ws, size_t ws_size,
                              hipStream_t stream)
{
    const float* x    = (const float*)d_in[0];
    const int*   eidx = (const int*)  d_in[1];
    const float* eij  = (const float*)d_in[2];
    const float* fW1  = (const float*)d_in[3];
    const float* fb1  = (const float*)d_in[4];
    const float* fW2  = (const float*)d_in[5];
    const float* fb2  = (const float*)d_in[6];
    const float* uW1  = (const float*)d_in[7];
    const float* ub1  = (const float*)d_in[8];
    const float* uW2  = (const float*)d_in[9];
    const float* ub2  = (const float*)d_in[10];

    const int N = in_sizes[0] / D_MODEL;
    const int E = in_sizes[2] / N_RBF;

    float* m = (float*)d_ws;                             // [N][128] accumulator
    hipMemsetAsync(m, 0, (size_t)N * D_MODEL * sizeof(float), stream);

    const int eblocks = (E + 63) / 64;
    edge_filter_scatter_kernel<<<eblocks, 128, 0, stream>>>(
        x, eidx, eij, fW1, fb1, fW2, fb2, m, N, E);

    const int nblocks = (N + 63) / 64;
    node_update_kernel<<<nblocks, 128, 0, stream>>>(
        x, m, uW1, ub1, uW2, ub2, (float*)d_out, N);
}
